// Model_39676907885507
// MI455X (gfx1250) — matrix operand order repaired and checked by reading, not compiled, not run on hardware
//
#include <hip/hip_runtime.h>

typedef __attribute__((ext_vector_type(16))) __bf16 v16bf;
typedef __attribute__((ext_vector_type(8)))  float  v8f;
typedef __attribute__((ext_vector_type(4)))  unsigned v4u;
typedef __attribute__((ext_vector_type(8)))  int      v8i;
typedef __attribute__((ext_vector_type(4)))  int      v4i;

constexpr int kB  = 4;     // batch
constexpr int kS  = 4096;  // sequence
constexpr int kD  = 128;   // head dim
constexpr int kTS = 64;    // key tile
constexpr int kWaves = 4;  // waves per block (wave32)
constexpr int kNT = kS / kTS;                 // 64 tiles per batch row-block
constexpr size_t kTileBytes = (size_t)kTS * kD * 2;  // 16384 B per pre-swizzled tile

// pack two f32 -> packed bf16x2 (low = first arg)
__device__ inline unsigned pk2(float a, float b) {
  union { __bf16 h[2]; unsigned u; } x;
  x.h[0] = (__bf16)a; x.h[1] = (__bf16)b;
  return x.u;
}

// exchange value with lane^16 (ds_swizzle group-of-32: xor=0x10, and=0x1f)
__device__ inline float swzx16f(float x) {
  int i = __builtin_bit_cast(int, x);
  i = __builtin_amdgcn_ds_swizzle(i, 0x401F);
  return __builtin_bit_cast(float, i);
}
__device__ inline unsigned swzx16u(unsigned x) {
  return (unsigned)__builtin_amdgcn_ds_swizzle((int)x, 0x401F);
}

__device__ inline v8f wmma_bf16(v16bf a, v16bf b, v8f c) {
  return __builtin_amdgcn_wmma_f32_16x16x32_bf16(false, a, false, b,
                                                 (short)0, c, false, false);
}

// byte offset of a __shared__ object within the workgroup LDS allocation
__device__ inline unsigned lds_offset(const void* p) {
  return (unsigned)(size_t)(__attribute__((address_space(3))) const void*)p;
}

// TDM: contiguous copy of 4096 dwords (16KB) global -> LDS (D# per ISA §8)
__device__ inline void tdm_copy16k(unsigned lds_byte, unsigned long long gaddr) {
  v4u g0;
  g0[0] = 1u;                                          // count=1
  g0[1] = lds_byte;                                    // lds_addr
  g0[2] = (unsigned)(gaddr & 0xFFFFFFFFull);           // global_addr[31:0]
  g0[3] = (unsigned)((gaddr >> 32) & 0x1FFFFFFull) | (2u << 30); // addr[56:32] | type=2
  // data_size=4B; tensor_dim0=4096, tensor_dim1=1; tile_dim0=4096, tile_dim1=1;
  // tensor_dim0_stride=4096; everything else zero.
  v8i g1 = { 0x00020000, 0x10000000, 0x00010000, 0x10000000, 1, 0x1000, 0, 0 };
  v4i z4 = { 0, 0, 0, 0 };
  v8i z8 = { 0, 0, 0, 0, 0, 0, 0, 0 };
  __builtin_amdgcn_tensor_load_to_lds(g0, g1, z4, z4, z8, 0);
}

// ---------------- pre-pass: fp32 K/V -> bf16, pre-swizzled A-fragment images ---
__global__ __launch_bounds__(256)
void prepack(const float* __restrict__ kin, const float* __restrict__ vin,
             unsigned* __restrict__ kpre, unsigned* __restrict__ vpre) {
  const size_t idx = (size_t)blockIdx.x * blockDim.x + threadIdx.x;
  const size_t nK = (size_t)kB * kS * (kD / 2);
  if (idx < nK) {
    // K: pack d-pair (d0,d0+1) of one key row into A-layout image
    const int dp = (int)(idx & (kD / 2 - 1));
    const size_t s = idx >> 6;                 // b*kS + row
    const int d0 = dp * 2;
    const int b = (int)(s / kS);
    const int row = (int)(s % kS);
    const int tile = row / kTS, r = row % kTS;
    const float* src = kin + s * kD + d0;
    const int grp = r >> 4, ll = r & 15;
    const int dl = d0 & 31, chunk = d0 >> 5;
    const int hb  = (dl >> 3) & 1;
    const int pos = (dl & 7) + 8 * ((dl >> 4) & 1);
    const size_t off = ((size_t)(b * kNT + tile)) * 4096 +
                       (size_t)(((grp * 4 + chunk) * 32) + (ll + 16 * hb)) * 8 + (pos >> 1);
    kpre[off] = pk2(src[0], src[1]);
  } else if (idx < 2 * nK) {
    // V: pack key-pair (s0,s0+1) at one d into V^T A-layout image
    const size_t j = idx - nK;
    const int d = (int)(j & (kD - 1));
    const size_t sp = j >> 7;                  // key-pair index
    const int b = (int)(sp / (kS / 2));
    const int rp = (int)(sp % (kS / 2));
    const int s0 = rp * 2;
    const int tile = s0 / kTS, kk0 = s0 % kTS;
    const float* src = vin + ((size_t)b * kS + s0) * kD + d;
    const int blk = kk0 >> 5, kkl = kk0 & 31;
    const int lane = (d & 15) + 16 * (kkl >> 4);
    const int pos = kkl & 15;
    const int dchunk = d >> 4;
    const size_t off = ((size_t)(b * kNT + tile)) * 4096 +
                       (size_t)(((blk * 8 + dchunk) * 32) + lane) * 8 + (pos >> 1);
    vpre[off] = pk2(src[0], src[kD]);
  }
}

// ---------------- main: TDM-fed, double-buffered flash attention ---------------
__global__ __launch_bounds__(kWaves * 32)
void fa_wmma_tdm(const float* __restrict__ q,
                 const unsigned* __restrict__ kpre,
                 const unsigned* __restrict__ vpre,
                 float* __restrict__ out) {
  __shared__ alignas(64) __bf16 KaBuf[2][4 * 4 * 32 * 16];   // 2 x 16KB
  __shared__ alignas(64) __bf16 VaBuf[2][2 * 8 * 32 * 16];   // 2 x 16KB

  const int tid  = threadIdx.x;
  const int w    = tid >> 5;
  const int lane = tid & 31;
  const int hi   = lane >> 4;
  const int ln   = lane & 15;

  const int qtilesPerBatch = kS / (16 * kWaves);
  const int b     = blockIdx.x / qtilesPerBatch;
  const int qt    = blockIdx.x % qtilesPerBatch;
  const int qrow0 = qt * (16 * kWaves) + w * 16;

  const float qscale = 1.44269504088896340736f * 0.08838834764831845f; // log2e/sqrt(128)

  // ---- Q resident in registers as four 32x16 B-fragments ----
  v16bf qB[4];
  {
    const float* qrow = q + ((size_t)b * kS + qrow0 + ln) * kD;
#pragma unroll
    for (int c = 0; c < 4; ++c) {
      float f[16];
      __builtin_memcpy(f, qrow + c * 32 + 8 * hi, 32);
      __builtin_memcpy(f + 8, qrow + c * 32 + 16 + 8 * hi, 32);
#pragma unroll
      for (int i = 0; i < 16; ++i) qB[c][i] = (__bf16)(f[i] * qscale);
    }
  }

  v8f acc[8];
#pragma unroll
  for (int d = 0; d < 8; ++d) acc[d] = v8f{};
  float mrow = -__builtin_inff();
  float lrow = 0.0f;

  const unsigned long long kbase =
      (unsigned long long)(size_t)kpre + (size_t)b * kNT * kTileBytes;
  const unsigned long long vbase =
      (unsigned long long)(size_t)vpre + (size_t)b * kNT * kTileBytes;
  const unsigned kaOff[2] = { lds_offset(&KaBuf[0][0]), lds_offset(&KaBuf[1][0]) };
  const unsigned vaOff[2] = { lds_offset(&VaBuf[0][0]), lds_offset(&VaBuf[1][0]) };

  // prologue: DMA tile 0 into buffer 0
  if (w == 0) {
    tdm_copy16k(kaOff[0], kbase);
    tdm_copy16k(vaOff[0], vbase);
    __builtin_amdgcn_s_wait_tensorcnt(0);
  }
  __syncthreads();

  for (int t = 0; t < kNT; ++t) {
    const int cur = t & 1;
    if (w == 0 && t + 1 < kNT) {   // prefetch next tile into other buffer
      tdm_copy16k(kaOff[cur ^ 1], kbase + (size_t)(t + 1) * kTileBytes);
      tdm_copy16k(vaOff[cur ^ 1], vbase + (size_t)(t + 1) * kTileBytes);
    }
    const __bf16* Ka = KaBuf[cur];
    const __bf16* Va = VaBuf[cur];

    // ---- scores: S^T = K Q^T, four 16x16 frags, K-dim chained ----
    v8f s[4];
#pragma unroll
    for (int g4 = 0; g4 < 4; ++g4) s[g4] = v8f{};
#pragma unroll
    for (int c = 0; c < 4; ++c) {
#pragma unroll
      for (int g4 = 0; g4 < 4; ++g4) {
        v16bf aK;
        __builtin_memcpy(&aK, &Ka[((g4 * 4 + c) * 32 + lane) * 16], 32);
        s[g4] = wmma_bf16(aK, qB[c], s[g4]);
      }
    }

    // ---- per-lane scalar online softmax (lane = query column) ----
    float mx = s[0][0];
#pragma unroll
    for (int g4 = 0; g4 < 4; ++g4)
#pragma unroll
      for (int g = 0; g < 8; ++g) mx = fmaxf(mx, s[g4][g]);
    mx = fmaxf(mx, swzx16f(mx));
    const float mn   = fmaxf(mrow, mx);
    const float corr = __builtin_amdgcn_exp2f(mrow - mn);
    float p[4][8];
    float rs = 0.0f;
#pragma unroll
    for (int g4 = 0; g4 < 4; ++g4)
#pragma unroll
      for (int g = 0; g < 8; ++g) {
        p[g4][g] = __builtin_amdgcn_exp2f(s[g4][g] - mn);
        rs += p[g4][g];
      }
    rs += swzx16f(rs);
    lrow = lrow * corr + rs;
    mrow = mn;
#pragma unroll
    for (int d = 0; d < 8; ++d)
#pragma unroll
      for (int g = 0; g < 8; ++g) acc[d][g] *= corr;

    // ---- O^T += V^T P^T : build P^T B-frags in-register (xor16 exchange) ----
#pragma unroll
    for (int kb = 0; kb < 2; ++kb) {
      unsigned a4[4], b4[4];
#pragma unroll
      for (int i = 0; i < 4; ++i) {
        a4[i] = pk2(p[2 * kb][2 * i],     p[2 * kb][2 * i + 1]);
        b4[i] = pk2(p[2 * kb + 1][2 * i], p[2 * kb + 1][2 * i + 1]);
      }
      unsigned vbu[8];
#pragma unroll
      for (int i = 0; i < 4; ++i) {
        const unsigned sa = swzx16u(a4[i]);
        const unsigned sb = swzx16u(b4[i]);
        vbu[i]     = hi ? sb    : a4[i];
        vbu[4 + i] = hi ? b4[i] : sa;
      }
      v16bf pB;
      __builtin_memcpy(&pB, vbu, 32);
#pragma unroll
      for (int dc = 0; dc < 8; ++dc) {
        v16bf aV;
        __builtin_memcpy(&aV, &Va[((kb * 8 + dc) * 32 + lane) * 16], 32);
        acc[dc] = wmma_bf16(aV, pB, acc[dc]);
      }
    }

    if (t + 1 < kNT) {
      if (w == 0) __builtin_amdgcn_s_wait_tensorcnt(0);
      __syncthreads();   // next tile ready AND current buffers free
    }
  }

  const float inv = 1.0f / lrow;
  float* orow = out + ((size_t)b * kS + qrow0 + ln) * kD + 8 * hi;
#pragma unroll
  for (int dc = 0; dc < 8; ++dc) {
    float o8[8];
#pragma unroll
    for (int g = 0; g < 8; ++g) o8[g] = acc[dc][g] * inv;
    __builtin_memcpy(orow + dc * 16, o8, 32);
  }
}

// ---------------- fallback (round-3 style, no workspace needed) ----------------
__global__ __launch_bounds__(kWaves * 32)
void fa_wmma_fb(const float* __restrict__ q, const float* __restrict__ k,
                const float* __restrict__ v, float* __restrict__ out) {
  __shared__ alignas(32) __bf16 Ka[4 * 4 * 32 * 16];
  __shared__ alignas(32) __bf16 Va[2 * 8 * 32 * 16];
  const int tid = threadIdx.x, w = tid >> 5, lane = tid & 31, hi = lane >> 4, ln = lane & 15;
  const int qtpb = kS / (16 * kWaves);
  const int b = blockIdx.x / qtpb, qt = blockIdx.x % qtpb;
  const int qrow0 = qt * (16 * kWaves) + w * 16;
  const float qscale = 1.44269504088896340736f * 0.08838834764831845f;
  v16bf qB[4];
  {
    const float* qrow = q + ((size_t)b * kS + qrow0 + ln) * kD;
#pragma unroll
    for (int c = 0; c < 4; ++c) {
      float f[16]; __builtin_memcpy(f, qrow + c * 32 + 8 * hi, 32); __builtin_memcpy(f + 8, qrow + c * 32 + 16 + 8 * hi, 32);
#pragma unroll
      for (int i = 0; i < 16; ++i) qB[c][i] = (__bf16)(f[i] * qscale);
    }
  }
  v8f acc[8];
#pragma unroll
  for (int d = 0; d < 8; ++d) acc[d] = v8f{};
  float mrow = -__builtin_inff(), lrow = 0.0f;
  const int kp = tid >> 2, k0 = kp * 2, d0 = (tid & 3) * 32;
  const int blkV = k0 >> 5, kkV = k0 & 31, halfV = kkV >> 4;
  const int idxV = kkV & 15;
  for (int t = 0; t < kNT; ++t) {
    __syncthreads();
    {
      const float* kg = k + ((size_t)b * kS + t * kTS + k0) * kD + d0;
      const float* vg = v + ((size_t)b * kS + t * kTS + k0) * kD + d0;
      const int chunkK = d0 >> 5;
#pragma unroll
      for (int r = 0; r < 2; ++r) {
        float src[32]; __builtin_memcpy(src, kg + r * kD, 128);
        const int key = k0 + r, grp = key >> 4, ll = key & 15;
        unsigned lo8[8], hi8[8];
#pragma unroll
        for (int i = 0; i < 4; ++i) {
          lo8[i] = pk2(src[2 * i], src[2 * i + 1]);
          lo8[4 + i] = pk2(src[16 + 2 * i], src[17 + 2 * i]);
          hi8[i] = pk2(src[8 + 2 * i], src[9 + 2 * i]);
          hi8[4 + i] = pk2(src[24 + 2 * i], src[25 + 2 * i]);
        }
        __bf16* base = &Ka[((grp * 4 + chunkK) * 32) * 16];
        __builtin_memcpy(base + (size_t)ll * 16, lo8, 32);
        __builtin_memcpy(base + (size_t)(16 + ll) * 16, hi8, 32);
      }
      float v0r[32], v1r[32];
      __builtin_memcpy(v0r, vg, 128);
      __builtin_memcpy(v1r, vg + kD, 128);
#pragma unroll
      for (int c = 0; c < 2; ++c) {
        const int chunk = (d0 >> 4) + c;
#pragma unroll
        for (int i = 0; i < 16; ++i) {
          unsigned u = pk2(v0r[c * 16 + i], v1r[c * 16 + i]);
          __builtin_memcpy(&Va[(((blkV * 8 + chunk) * 32) + i + 16 * halfV) * 16 + idxV], &u, 4);
        }
      }
    }
    __syncthreads();
    v8f s[4];
#pragma unroll
    for (int g4 = 0; g4 < 4; ++g4) s[g4] = v8f{};
#pragma unroll
    for (int c = 0; c < 4; ++c)
#pragma unroll
      for (int g4 = 0; g4 < 4; ++g4) {
        v16bf aK; __builtin_memcpy(&aK, &Ka[((g4 * 4 + c) * 32 + lane) * 16], 32);
        s[g4] = wmma_bf16(aK, qB[c], s[g4]);
      }
    float mx = s[0][0];
#pragma unroll
    for (int g4 = 0; g4 < 4; ++g4)
#pragma unroll
      for (int g = 0; g < 8; ++g) mx = fmaxf(mx, s[g4][g]);
    mx = fmaxf(mx, swzx16f(mx));
    const float mn = fmaxf(mrow, mx);
    const float corr = __builtin_amdgcn_exp2f(mrow - mn);
    float p[4][8];
    float rs = 0.0f;
#pragma unroll
    for (int g4 = 0; g4 < 4; ++g4)
#pragma unroll
      for (int g = 0; g < 8; ++g) { p[g4][g] = __builtin_amdgcn_exp2f(s[g4][g] - mn); rs += p[g4][g]; }
    rs += swzx16f(rs);
    lrow = lrow * corr + rs; mrow = mn;
#pragma unroll
    for (int d = 0; d < 8; ++d)
#pragma unroll
      for (int g = 0; g < 8; ++g) acc[d][g] *= corr;
#pragma unroll
    for (int kb = 0; kb < 2; ++kb) {
      unsigned a4[4], b4[4];
#pragma unroll
      for (int i = 0; i < 4; ++i) {
        a4[i] = pk2(p[2 * kb][2 * i], p[2 * kb][2 * i + 1]);
        b4[i] = pk2(p[2 * kb + 1][2 * i], p[2 * kb + 1][2 * i + 1]);
      }
      unsigned vbu[8];
#pragma unroll
      for (int i = 0; i < 4; ++i) {
        const unsigned sa = swzx16u(a4[i]);
        const unsigned sb = swzx16u(b4[i]);
        vbu[i] = hi ? sb : a4[i];
        vbu[4 + i] = hi ? b4[i] : sa;
      }
      v16bf pB; __builtin_memcpy(&pB, vbu, 32);
#pragma unroll
      for (int dc = 0; dc < 8; ++dc) {
        v16bf aV; __builtin_memcpy(&aV, &Va[((kb * 8 + dc) * 32 + lane) * 16], 32);
        acc[dc] = wmma_bf16(aV, pB, acc[dc]);
      }
    }
  }
  const float inv = 1.0f / lrow;
  float* orow = out + ((size_t)b * kS + qrow0 + ln) * kD + 8 * hi;
#pragma unroll
  for (int dc = 0; dc < 8; ++dc) {
    float o8[8];
#pragma unroll
    for (int g = 0; g < 8; ++g) o8[g] = acc[dc][g] * inv;
    __builtin_memcpy(orow + dc * 16, o8, 32);
  }
}

extern "C" void kernel_launch(void* const* d_in, const int* in_sizes, int n_in,
                              void* d_out, int out_size, void* d_ws, size_t ws_size,
                              hipStream_t stream) {
  const float* q = (const float*)d_in[0];
  const float* k = (const float*)d_in[1];
  const float* v = (const float*)d_in[2];
  float* out = (float*)d_out;
  const size_t preBytes = (size_t)kB * kS * kD * 2;   // 4MB per matrix
  dim3 grid(kB * (kS / (16 * kWaves)));
  dim3 block(kWaves * 32);
  if (ws_size >= 2 * preBytes) {
    unsigned* kpre = (unsigned*)d_ws;
    unsigned* vpre = (unsigned*)((char*)d_ws + preBytes);
    const size_t nOut = (size_t)kB * kS * kD;         // dwords for K+V combined
    prepack<<<dim3((unsigned)(nOut / 256)), dim3(256), 0, stream>>>(k, v, kpre, vpre);
    fa_wmma_tdm<<<grid, block, 0, stream>>>(q, kpre, vpre, out);
  } else {
    fa_wmma_fb<<<grid, block, 0, stream>>>(q, k, v, out);
  }
}
